// MaskedIPAttnProcessor2_0_26079041421772
// MI455X (gfx1250) — matrix operand order repaired and checked by reading, not compiled, not run on hardware
//
#include <hip/hip_runtime.h>
#include <hip/hip_bf16.h>

typedef __attribute__((ext_vector_type(16))) __bf16 bf16x16;
typedef __attribute__((ext_vector_type(8)))  __bf16 bf16x8;
typedef __attribute__((ext_vector_type(8)))  float  f32x8;

#define DEV static __device__ __forceinline__

DEV __bf16 f2bf(float f) {
  unsigned u = __float_as_uint(f);
  u += 0x7FFFu + ((u >> 16) & 1u);
  unsigned short h = (unsigned short)(u >> 16);
  return __builtin_bit_cast(__bf16, h);
}

DEV f32x8 wmma_bf16(bf16x16 a, bf16x16 b, f32x8 c) {
  return __builtin_amdgcn_wmma_f32_16x16x32_bf16(false, a, false, b, (short)0, c, false, false);
}

// low 32 bits of a generic pointer to LDS == wave-relative LDS byte offset (ISA 10.2)
DEV unsigned ldsAddr(const void* p) { return (unsigned)(unsigned long long)p; }

// async global->LDS copy, 16B per active lane; tracked by ASYNCcnt
DEV void asyncLdB128(unsigned lds, const void* g) {
  asm volatile("global_load_async_to_lds_b128 %0, %1, off" :: "v"(lds), "v"(g) : "memory");
}
DEV void waitAsync() { asm volatile("s_wait_asynccnt 0" ::: "memory"); }

// ---------------------------------------------------------------- weight transpose+cast
// src: (Kdim, Ndim) f32 row-major  ->  dst: (Ndim, Kdim) bf16 row-major
__global__ __launch_bounds__(256) void cast_transpose(const float* __restrict__ src,
                                                      __bf16* __restrict__ dst,
                                                      int Kdim, int Ndim) {
  __shared__ __bf16 tile[32][33];
  const int kt = blockIdx.x * 32, nt = blockIdx.y * 32;
  const int tx = threadIdx.x & 31, ty = threadIdx.x >> 5;  // 32 x 8
  #pragma unroll
  for (int r = ty; r < 32; r += 8)
    tile[r][tx] = f2bf(src[(long long)(kt + r) * Ndim + nt + tx]);
  __syncthreads();
  #pragma unroll
  for (int r = ty; r < 32; r += 8)
    dst[(long long)(nt + r) * Kdim + kt + tx] = tile[tx][r];
}

// ---------------------------------------------------------------- A-tile staging (16 elems/thread)
DEV void stageA16(const float* __restrict__ ap, __bf16* __restrict__ dst) {
  float4 f0 = *(const float4*)(ap + 0);
  float4 f1 = *(const float4*)(ap + 4);
  float4 f2 = *(const float4*)(ap + 8);
  float4 f3 = *(const float4*)(ap + 12);
  bf16x8 h0, h1;
  h0[0] = f2bf(f0.x); h0[1] = f2bf(f0.y); h0[2] = f2bf(f0.z); h0[3] = f2bf(f0.w);
  h0[4] = f2bf(f1.x); h0[5] = f2bf(f1.y); h0[6] = f2bf(f1.z); h0[7] = f2bf(f1.w);
  h1[0] = f2bf(f2.x); h1[1] = f2bf(f2.y); h1[2] = f2bf(f2.z); h1[3] = f2bf(f2.w);
  h1[4] = f2bf(f3.x); h1[5] = f2bf(f3.y); h1[6] = f2bf(f3.z); h1[7] = f2bf(f3.w);
  *(bf16x8*)dst       = h0;
  *(bf16x8*)(dst + 8) = h1;
}
DEV void stageA16(const __bf16* __restrict__ ap, __bf16* __restrict__ dst) {
  *(bf16x8*)dst       = *(const bf16x8*)ap;
  *(bf16x8*)(dst + 8) = *(const bf16x8*)(ap + 8);
}

// ---------------------------------------------------------------- GEMM (bf16 WMMA, f32 accum)
// C[M,N] = A[M,K] @ Bt[N,K]^T (+bias) (+residual); N mult of 64, K mult of 32.
// Bt is the pre-transposed bf16 weight, so B staging is pure b128 copies.
// Block: 256 thr = 8 waves (4x2), block tile 128x64, wave tile 32x32 (4 WMMA/step).
// Double-buffered LDS, one barrier per K-step, global_prefetch on next tiles.
template <typename TA>
__global__ __launch_bounds__(256) void gemm_wmma(
    const TA* __restrict__ A, long long sA,
    const __bf16* __restrict__ Bt,
    void* __restrict__ Cv, long long sC,
    int M, int N, int K,
    const float* __restrict__ bias,
    const float* __restrict__ residual, long long sR,
    int storeBf16)
{
  __shared__ __align__(16) __bf16 As[2][128 * 32];
  __shared__ __align__(16) __bf16 Bs[2][64 * 32];   // [n][k]

  const int t = threadIdx.x, lane = t & 31, wave = t >> 5;
  const int wm = wave >> 1, wn = wave & 1;          // 4 waves (M) x 2 waves (N)
  const int m0 = blockIdx.x * 128, n0 = blockIdx.y * 64;
  const int z = blockIdx.z;
  A += (long long)z * sA;

  const int hi  = (lane < 16) ? 0 : 1;
  const int l15 = lane & 15;

  // staging assignments
  const int ar  = t >> 1;           // 0..127 (A row in tile)
  const int ac  = (t & 1) * 16;     // 0 / 16 (A col half)
  int mr = m0 + ar; if (mr > M - 1) mr = M - 1;
  const TA* ag = &A[(long long)mr * K + ac];
  const int bn  = t >> 2;           // 0..63  (B n row of Bt)
  const int bkk = (t & 3) * 8;      // 0..24  (B k chunk)
  const __bf16* bg = &Bt[(long long)(n0 + bn) * K + bkk];

  f32x8 acc[2][2] = {};

  int p = 0;
  for (int k0 = 0; k0 < K; k0 += 32, p ^= 1) {
    if (k0 + 32 < K) {
      __builtin_prefetch(ag + k0 + 32, 0, 1);
      __builtin_prefetch(bg + k0 + 32, 0, 1);
    }
    // stage A (convert f32->bf16 if needed)
    stageA16(ag + k0, &As[p][ar * 32 + ac]);
    // stage B (already transposed in global): contiguous 16B copy
    *(bf16x8*)&Bs[p][bn * 32 + bkk] = *(const bf16x8*)(bg + k0);
    __syncthreads();

    const __bf16* arp0 = &As[p][(wm * 32 + l15) * 32];
    const __bf16* arp1 = arp0 + 16 * 32;
    bf16x16 a0, a1, b0, b1;
    *(bf16x8*)&a0       = *(const bf16x8*)(arp0 + hi * 8);
    *((bf16x8*)&a0 + 1) = *(const bf16x8*)(arp0 + 16 + hi * 8);
    *(bf16x8*)&a1       = *(const bf16x8*)(arp1 + hi * 8);
    *((bf16x8*)&a1 + 1) = *(const bf16x8*)(arp1 + 16 + hi * 8);
    const __bf16* brp0 = &Bs[p][(wn * 32 + l15) * 32];
    const __bf16* brp1 = brp0 + 16 * 32;
    *(bf16x8*)&b0       = *(const bf16x8*)(brp0 + hi * 8);
    *((bf16x8*)&b0 + 1) = *(const bf16x8*)(brp0 + 16 + hi * 8);
    *(bf16x8*)&b1       = *(const bf16x8*)(brp1 + hi * 8);
    *((bf16x8*)&b1 + 1) = *(const bf16x8*)(brp1 + 16 + hi * 8);

    acc[0][0] = wmma_bf16(a0, b0, acc[0][0]);
    acc[0][1] = wmma_bf16(a0, b1, acc[0][1]);
    acc[1][0] = wmma_bf16(a1, b0, acc[1][0]);
    acc[1][1] = wmma_bf16(a1, b1, acc[1][1]);
    // no trailing barrier: next iteration writes the other buffer, and the
    // barrier's dscnt-0 wait covers this iteration's LDS reads.
  }

  #pragma unroll
  for (int im = 0; im < 2; ++im)
    #pragma unroll
    for (int jn = 0; jn < 2; ++jn) {
      int col = n0 + wn * 32 + jn * 16 + l15;
      #pragma unroll
      for (int i = 0; i < 8; ++i) {
        int row = m0 + wm * 32 + im * 16 + i + hi * 8;
        if (row < M) {
          float v = acc[im][jn][i];
          if (bias)     v += bias[col];
          if (residual) v += residual[(long long)z * sR + (long long)row * N + col];
          if (storeBf16) ((__bf16*)Cv)[(long long)z * sC + (long long)row * N + col] = f2bf(v);
          else           ((float*) Cv)[(long long)z * sC + (long long)row * N + col] = v;
        }
      }
    }
}

// ---------------------------------------------------------------- fused masked-IP attention
// grid = (S/128, B*H); block = 256 threads (8 waves, one 16-row q tile each)
__global__ __launch_bounds__(256) void attn_kernel(
    const __bf16* __restrict__ Q,    // (B,S,D)
    const __bf16* __restrict__ Kt,   // (B,77,D)
    const __bf16* __restrict__ Vt,   // (B,77,D)
    const __bf16* __restrict__ Kip,  // (B,8,D)
    const __bf16* __restrict__ Vip,  // (B,8,D)
    const float* __restrict__ bbox,  // (B,4,4)
    __bf16* __restrict__ Out)        // (B,S,D)
{
  constexpr int S = 4096, D = 1280, H = 20, DH = 64, TL = 77, KP = 96;
  __shared__ __align__(16) __bf16 kl [KP * DH];      // [kv][dh], kv 77..95 zero
  __shared__ __align__(16) __bf16 vtl[DH * KP];      // [dh][kv], kv 77..95 zero
  __shared__ __align__(16) __bf16 kip[16 * DH];      // [kv][dh], kv 8..15 zero
  __shared__ __align__(16) __bf16 vip[DH * 32];      // [dh][kv], kv 8..31 zero
  __shared__ __align__(16) __bf16 pw [8 * 16 * KP];  // per-wave prob staging (16 x 96)

  const int t = threadIdx.x, lane = t & 31, wave = t >> 5;
  const int bh = blockIdx.y, b = bh / H, h = bh % H;
  const int l15 = lane & 15;
  const int hi  = (lane < 16) ? 0 : 1;
  const __bf16 bz = f2bf(0.f);

  // ---- stage K / K_ip via async global->LDS; V^T / V_ip^T via scalar transpose
  {
    const __bf16* kg = Kt + (long long)b * TL * D + h * DH;
    for (int c = t; c < KP * 8; c += 256) {
      int r = c >> 3, sg = (c & 7) * 8;
      if (r < TL) asyncLdB128(ldsAddr(&kl[r * DH + sg]), &kg[(long long)r * D + sg]);
      else        *(bf16x8*)&kl[r * DH + sg] = (bf16x8){};
    }
    const __bf16* kig = Kip + (long long)b * 8 * D + h * DH;
    for (int c = t; c < 16 * 8; c += 256) {
      int r = c >> 3, sg = (c & 7) * 8;
      if (r < 8) asyncLdB128(ldsAddr(&kip[r * DH + sg]), &kig[(long long)r * D + sg]);
      else       *(bf16x8*)&kip[r * DH + sg] = (bf16x8){};
    }
    const __bf16* vg = Vt + (long long)b * TL * D + h * DH;
    for (int c = t; c < DH * KP; c += 256) {
      int d = c / KP, kv = c % KP;
      vtl[d * KP + kv] = (kv < TL) ? vg[(long long)kv * D + d] : bz;
    }
    const __bf16* vig = Vip + (long long)b * 8 * D + h * DH;
    for (int c = t; c < DH * 32; c += 256) {
      int d = c >> 5, kv = c & 31;
      vip[d * 32 + kv] = (kv < 8) ? vig[(long long)kv * D + d] : bz;
    }
  }
  waitAsync();       // ASYNCcnt==0: async LDS writes complete (barrier only covers DScnt)
  __syncthreads();

  // ---- load q A-fragments (dh=64 -> two K=32 chunks), reused for text + ip scores
  const int s0 = (blockIdx.x * 8 + wave) * 16;
  const __bf16* qg = Q + ((long long)b * S + s0) * D + h * DH;
  const __bf16* qr = qg + (long long)l15 * D;
  bf16x16 qa0, qa1;
  *(bf16x8*)&qa0       = *(const bf16x8*)(qr + hi * 16);
  *((bf16x8*)&qa0 + 1) = *(const bf16x8*)(qr + hi * 16 + 8);
  *(bf16x8*)&qa1       = *(const bf16x8*)(qr + 32 + hi * 16);
  *((bf16x8*)&qa1 + 1) = *(const bf16x8*)(qr + 32 + hi * 16 + 8);

  const int kb = hi * 16;

  // ---- text scores: 5 kv tiles (0..79), K=64 -> 2 WMMA each
  f32x8 sc[5];
  #pragma unroll
  for (int tt = 0; tt < 5; ++tt) {
    f32x8 a = {};
    const __bf16* br = &kl[(tt * 16 + l15) * DH];
    bf16x16 b0, b1;
    *(bf16x8*)&b0       = *(const bf16x8*)(br + kb);
    *((bf16x8*)&b0 + 1) = *(const bf16x8*)(br + kb + 8);
    *(bf16x8*)&b1       = *(const bf16x8*)(br + 32 + kb);
    *((bf16x8*)&b1 + 1) = *(const bf16x8*)(br + 32 + kb + 8);
    a = wmma_bf16(qa0, b0, a);
    a = wmma_bf16(qa1, b1, a);
    sc[tt] = a;
  }

  // ---- text softmax (slot i -> row i / i+8; cols across 16-lane half-wave)
  const float SCALE = 0.125f, L2E = 1.44269504088896f;
  #pragma unroll
  for (int i = 0; i < 8; ++i) {
    float mx = -1e30f;
    #pragma unroll
    for (int tt = 0; tt < 5; ++tt) {
      float s = sc[tt][i] * SCALE;
      if (tt * 16 + l15 >= TL) s = -1e30f;
      sc[tt][i] = s;
      mx = fmaxf(mx, s);
    }
    #pragma unroll
    for (int m = 8; m >= 1; m >>= 1) mx = fmaxf(mx, __shfl_xor(mx, m, 32));
    float sum = 0.f;
    #pragma unroll
    for (int tt = 0; tt < 5; ++tt) {
      float e = exp2f((sc[tt][i] - mx) * L2E);
      if (tt * 16 + l15 >= TL) e = 0.f;
      sc[tt][i] = e;
      sum += e;
    }
    #pragma unroll
    for (int m = 8; m >= 1; m >>= 1) sum += __shfl_xor(sum, m, 32);
    float inv = 1.f / sum;
    #pragma unroll
    for (int tt = 0; tt < 5; ++tt) sc[tt][i] *= inv;
  }

  // ---- stage text probs as bf16 A-matrix (16 x 96, cols >=77 zero)
  __bf16* pwv = &pw[wave * 16 * KP];
  #pragma unroll
  for (int i = 0; i < 8; ++i) {
    int row = i + hi * 8;
    #pragma unroll
    for (int tt = 0; tt < 6; ++tt) {
      float v = (tt < 5) ? sc[tt][i] : 0.f;
      pwv[row * KP + tt * 16 + l15] = f2bf(v);
    }
  }

  // ---- out = P @ V  (K=96 -> 3 chunks, 4 dh tiles)
  f32x8 outv[4] = {{}, {}, {}, {}};
  #pragma unroll
  for (int c = 0; c < 3; ++c) {
    bf16x16 pa;
    const __bf16* pr = pwv + l15 * KP + c * 32;
    *(bf16x8*)&pa       = *(const bf16x8*)(pr + hi * 16);
    *((bf16x8*)&pa + 1) = *(const bf16x8*)(pr + hi * 16 + 8);
    #pragma unroll
    for (int d = 0; d < 4; ++d) {
      const __bf16* vr = &vtl[(d * 16 + l15) * KP + c * 32];
      bf16x16 vb;
      *(bf16x8*)&vb       = *(const bf16x8*)(vr + kb);
      *((bf16x8*)&vb + 1) = *(const bf16x8*)(vr + kb + 8);
      outv[d] = wmma_bf16(pa, vb, outv[d]);
    }
  }

  // ---- ip scores (single kv tile, cols 0..7 valid)
  f32x8 sip = {};
  {
    const __bf16* br = &kip[l15 * DH];
    bf16x16 b0, b1;
    *(bf16x8*)&b0       = *(const bf16x8*)(br + kb);
    *((bf16x8*)&b0 + 1) = *(const bf16x8*)(br + kb + 8);
    *(bf16x8*)&b1       = *(const bf16x8*)(br + 32 + kb);
    *((bf16x8*)&b1 + 1) = *(const bf16x8*)(br + 32 + kb + 8);
    sip = wmma_bf16(qa0, b0, sip);
    sip = wmma_bf16(qa1, b1, sip);
  }

  // ---- bbox mask + ip softmax  (grid 64x64, linspace -> idx/63)
  const float* bb = bbox + b * 16;
  float bx1[4], by1[4], bx2[4], by2[4];
  #pragma unroll
  for (int m = 0; m < 4; ++m) {
    bx1[m] = bb[m * 4 + 0]; by1[m] = bb[m * 4 + 1];
    bx2[m] = bb[m * 4 + 2]; by2[m] = bb[m * 4 + 3];
  }
  #pragma unroll
  for (int i = 0; i < 8; ++i) {
    int srow = s0 + i + hi * 8;
    float gx = (float)(srow & 63) * (1.0f / 63.0f);
    float gy = (float)(srow >> 6) * (1.0f / 63.0f);
    bool any = false, ic = false;
    #pragma unroll
    for (int m = 0; m < 4; ++m) {
      bool ins = (gx >= bx1[m]) && (gx <= bx2[m]) && (gy >= by1[m]) && (gy <= by2[m]);
      any = any || ins;
      if (l15 == 4 + m) ic = ins;
    }
    float mk = (l15 < 4) ? (any ? -10000.f : 0.f) : (ic ? 0.f : -10000.f);
    float s = sip[i] * SCALE + mk;
    if (l15 >= 8) s = -1e30f;
    float mx = s;
    #pragma unroll
    for (int m = 8; m >= 1; m >>= 1) mx = fmaxf(mx, __shfl_xor(mx, m, 32));
    float e = exp2f((s - mx) * L2E);
    if (l15 >= 8) e = 0.f;
    float sum = e;
    #pragma unroll
    for (int m = 8; m >= 1; m >>= 1) sum += __shfl_xor(sum, m, 32);
    sip[i] = e / sum;
  }

  // ---- stage ip probs (16 x 32, cols >=8 zero) then accumulate P_ip @ V_ip
  #pragma unroll
  for (int i = 0; i < 8; ++i) {
    int row = i + hi * 8;
    pwv[row * KP + l15]      = f2bf(sip[i]);
    pwv[row * KP + 16 + l15] = bz;
  }
  {
    bf16x16 pa;
    const __bf16* pr = pwv + l15 * KP;
    *(bf16x8*)&pa       = *(const bf16x8*)(pr + hi * 16);
    *((bf16x8*)&pa + 1) = *(const bf16x8*)(pr + hi * 16 + 8);
    #pragma unroll
    for (int d = 0; d < 4; ++d) {
      const __bf16* vr = &vip[(d * 16 + l15) * 32];
      bf16x16 vb;
      *(bf16x8*)&vb       = *(const bf16x8*)(vr + kb);
      *((bf16x8*)&vb + 1) = *(const bf16x8*)(vr + kb + 8);
      outv[d] = wmma_bf16(pa, vb, outv[d]);  // IP_SCALE == 1
    }
  }

  // ---- store merged-head output as bf16
  __bf16* og = Out + ((long long)b * S + s0) * D + h * DH;
  #pragma unroll
  for (int d = 0; d < 4; ++d)
    #pragma unroll
    for (int i = 0; i < 8; ++i) {
      int row = i + hi * 8;
      og[(long long)row * D + d * 16 + l15] = f2bf(outv[d][i]);
    }
}

// ---------------------------------------------------------------- host side
template <typename TA>
static void launch_gemm(const void* A, long long sA, const void* Bt,
                        void* C, long long sC, int M, int N, int K, int batches,
                        const float* bias, const float* residual, long long sR,
                        int storeBf16, hipStream_t stream) {
  dim3 grid((M + 127) / 128, N / 64, batches);
  gemm_wmma<TA><<<grid, 256, 0, stream>>>((const TA*)A, sA, (const __bf16*)Bt,
                                          C, sC, M, N, K, bias, residual, sR, storeBf16);
}

extern "C" void kernel_launch(void* const* d_in, const int* in_sizes, int n_in,
                              void* d_out, int out_size, void* d_ws, size_t ws_size,
                              hipStream_t stream) {
  (void)in_sizes; (void)n_in; (void)out_size; (void)ws_size;
  const float* hidden = (const float*)d_in[0];   // (4,4096,1280)
  const float* enc    = (const float*)d_in[1];   // (4,85,2048)
  const float* bbox   = (const float*)d_in[2];   // (4,4,4)
  const float* w_q    = (const float*)d_in[3];
  const float* w_k    = (const float*)d_in[4];
  const float* w_v    = (const float*)d_in[5];
  const float* w_kip  = (const float*)d_in[6];
  const float* w_vip  = (const float*)d_in[7];
  const float* w_out  = (const float*)d_in[8];
  const float* b_out  = (const float*)d_in[9];

  const long long B = 4, S = 4096, D = 1280, C = 2048, TL = 77, IPN = 8, EL = 85;
  const long long MS = B * S;  // 16384

  size_t off = 0;
  char* wsb = (char*)d_ws;
  auto alloc = [&](long long elems) {
    __bf16* p = (__bf16*)(wsb + off);
    off += ((size_t)elems * 2 + 255) & ~(size_t)255;
    return p;
  };
  __bf16* wqt   = alloc(D * D);     // (D,D)   transposed bf16
  __bf16* wkt   = alloc(D * C);     // (D,C)
  __bf16* wvt   = alloc(D * C);
  __bf16* wkipt = alloc(D * C);
  __bf16* wvipt = alloc(D * C);
  __bf16* woutt = alloc(D * D);
  __bf16* qb    = alloc(MS * D);
  __bf16* kbuf  = alloc(B * TL * D);
  __bf16* vbuf  = alloc(B * TL * D);
  __bf16* kipb  = alloc(B * IPN * D);
  __bf16* vipb  = alloc(B * IPN * D);
  __bf16* attnb = alloc(MS * D);

  auto tcast = [&](const float* s, __bf16* d, long long Kd, long long Nd) {
    cast_transpose<<<dim3((unsigned)(Kd / 32), (unsigned)(Nd / 32)), 256, 0, stream>>>(
        s, d, (int)Kd, (int)Nd);
  };
  tcast(w_q,   wqt,   D, D);
  tcast(w_k,   wkt,   C, D);
  tcast(w_v,   wvt,   C, D);
  tcast(w_kip, wkipt, C, D);
  tcast(w_vip, wvipt, C, D);
  tcast(w_out, woutt, D, D);

  // projections (A read as f32 directly, converted while staging)
  launch_gemm<float>(hidden, 0, wqt, qb, 0, (int)MS, (int)D, (int)D, 1,
                     nullptr, nullptr, 0, 1, stream);
  launch_gemm<float>(enc, EL * C, wkt, kbuf, TL * D, (int)TL, (int)D, (int)C, (int)B,
                     nullptr, nullptr, 0, 1, stream);
  launch_gemm<float>(enc, EL * C, wvt, vbuf, TL * D, (int)TL, (int)D, (int)C, (int)B,
                     nullptr, nullptr, 0, 1, stream);
  launch_gemm<float>(enc + TL * C, EL * C, wkipt, kipb, IPN * D, (int)IPN, (int)D, (int)C, (int)B,
                     nullptr, nullptr, 0, 1, stream);
  launch_gemm<float>(enc + TL * C, EL * C, wvipt, vipb, IPN * D, (int)IPN, (int)D, (int)C, (int)B,
                     nullptr, nullptr, 0, 1, stream);

  // fused dual attention (text + masked IP)
  attn_kernel<<<dim3((unsigned)(S / 128), (unsigned)(B * 20)), 256, 0, stream>>>(
      qb, kbuf, vbuf, kipb, vipb, bbox, attnb);

  // output projection + bias + residual -> f32 d_out
  launch_gemm<__bf16>(attnb, 0, woutt, d_out, 0, (int)MS, (int)D, (int)D, 1,
                      b_out, hidden, 0, 0, stream);
}
